// MSAEEncoder_59433757442411
// MI455X (gfx1250) — compile-verified, matrix operand order checked by reading, not run on hardware
//
#include <hip/hip_runtime.h>
#include <hip/hip_bf16.h>

// ---------------------------------------------------------------------------
// MSAE encoder: h = x @ W + b  (16384x1024 @ 1024x4096, fp32)
// then for k in {32,64,128}: out_k = relu(h * topk_mask(h, k))
// Strategy:
//   Kernel 1: bf16 WMMA GEMM (fp32 accumulate), h -> out slice 0.
//   Kernel 2: per-row exact radix-select thresholds (k=32/64/128),
//             write 3 masked+relu slices (slice 0 rewritten in place).
// ---------------------------------------------------------------------------

typedef __attribute__((ext_vector_type(16))) __bf16       v16bf;
typedef __attribute__((ext_vector_type(8)))  __bf16       v8bf;
typedef __attribute__((ext_vector_type(2)))  __bf16       v2bf;
typedef __attribute__((ext_vector_type(8)))  float        v8f;
typedef __attribute__((ext_vector_type(4)))  unsigned int v4u;
typedef __attribute__((ext_vector_type(8)))  unsigned int v8u;

#define N_ROWS 16384
#define D_IN   1024
#define D_HID  4096
#define BM 128
#define BN 128
#define BK 32

// -------------------------------- GEMM -------------------------------------
__global__ __launch_bounds__(256)
void msae_gemm_bf16(const float* __restrict__ x, const float* __restrict__ W,
                    const float* __restrict__ bias, float* __restrict__ h)
{
    __shared__ __align__(16) __bf16 Ab[2][BM * BK];   // [m][k], 64B per row
    __shared__ __align__(16) __bf16 Bb[2][BN * BK];   // [n][k], 64B per col

    const int tid = threadIdx.x;
    const int m0  = blockIdx.y * BM;
    const int n0  = blockIdx.x * BN;

    // Stage one BKxBM / BKxBN tile pair: global fp32 -> bf16 -> LDS.
    auto stage = [&](int buf, int kt) {
        const int k0 = kt * BK;
        {   // A: 128 rows x 32 k. thread covers (row, half of K-16-chunk)
            const int row  = tid & 127;
            const int half = tid >> 7;                 // 0..1 -> k offset 0/16
            const float4* g =
                (const float4*)(x + (size_t)(m0 + row) * D_IN + k0 + half * 16);
            float4 f0 = g[0], f1 = g[1], f2 = g[2], f3 = g[3];
            v8bf p0, p1;
            p0[0] = (__bf16)f0.x; p0[1] = (__bf16)f0.y;
            p0[2] = (__bf16)f0.z; p0[3] = (__bf16)f0.w;
            p0[4] = (__bf16)f1.x; p0[5] = (__bf16)f1.y;
            p0[6] = (__bf16)f1.z; p0[7] = (__bf16)f1.w;
            p1[0] = (__bf16)f2.x; p1[1] = (__bf16)f2.y;
            p1[2] = (__bf16)f2.z; p1[3] = (__bf16)f2.w;
            p1[4] = (__bf16)f3.x; p1[5] = (__bf16)f3.y;
            p1[6] = (__bf16)f3.z; p1[7] = (__bf16)f3.w;
            __bf16* d = &Ab[buf][row * BK + half * 16];
            *(v8bf*)(d)     = p0;
            *(v8bf*)(d + 8) = p1;
        }
        {   // B: global row-major [k][n]; LDS column-major [n][k].
            // thread covers 2 consecutive k rows x 8 consecutive n cols.
            const int kk = (tid & 15) * 2;
            const int nb = (tid >> 4) * 8;
            const float* g0 = W + (size_t)(k0 + kk) * D_HID + n0 + nb;
            const float* g1 = g0 + D_HID;
            float4 r0a = ((const float4*)g0)[0], r0b = ((const float4*)g0)[1];
            float4 r1a = ((const float4*)g1)[0], r1b = ((const float4*)g1)[1];
            __bf16* base = &Bb[buf][kk];
#define PKB(nn, A, B)                                            \
            { v2bf p; p[0] = (__bf16)(A); p[1] = (__bf16)(B);    \
              *(v2bf*)(base + (size_t)(nb + (nn)) * BK) = p; }
            PKB(0, r0a.x, r1a.x); PKB(1, r0a.y, r1a.y);
            PKB(2, r0a.z, r1a.z); PKB(3, r0a.w, r1a.w);
            PKB(4, r0b.x, r1b.x); PKB(5, r0b.y, r1b.y);
            PKB(6, r0b.z, r1b.z); PKB(7, r0b.w, r1b.w);
#undef PKB
        }
    };

    // Fragment loader: 16-bit A/B layout, lanes 0-15: K0-7,16-23;
    // lanes 16-31: K8-15,24-31 (two ds_load_b128 per fragment).
    auto ldfrag = [&](const __bf16* base, int idx16, int lane) -> v16bf {
        const int r = lane & 15, sel = lane >> 4;
        const __bf16* p = base + (size_t)(idx16 * 16 + r) * BK + sel * 8;
        v4u lo = *(const v4u*)(p);        // K 0..7   (or 8..15)
        v4u hi = *(const v4u*)(p + 16);   // K 16..23 (or 24..31)
        v8u w  = __builtin_shufflevector(lo, hi, 0, 1, 2, 3, 4, 5, 6, 7);
        return __builtin_bit_cast(v16bf, w);
    };

    const int lane = tid & 31, wid = tid >> 5;
    const int wm = wid >> 2;   // 0..1, 64 rows each
    const int wn = wid & 3;    // 0..3, 32 cols each

    v8f zero = {0.f, 0.f, 0.f, 0.f, 0.f, 0.f, 0.f, 0.f};
    v8f acc[4][2];
#pragma unroll
    for (int i = 0; i < 4; ++i)
#pragma unroll
        for (int j = 0; j < 2; ++j) acc[i][j] = zero;

    const int KT = D_IN / BK;   // 32
    stage(0, 0);
    for (int kt = 0; kt < KT; ++kt) {
        __syncthreads();
        const int cur = kt & 1;
        if (kt + 2 < KT) {   // prefetch two K-tiles ahead -> global_prefetch_b8
            __builtin_prefetch(x + (size_t)(m0 + (tid & 127)) * D_IN
                                 + (size_t)(kt + 2) * BK, 0, 0);
            __builtin_prefetch(W + (size_t)((kt + 2) * BK + (tid & 31)) * D_HID
                                 + n0, 0, 0);
        }
        if (kt + 1 < KT) stage(cur ^ 1, kt + 1);

        v16bf af[4], bfrag[2];
#pragma unroll
        for (int i = 0; i < 4; ++i) af[i]    = ldfrag(Ab[cur], wm * 4 + i, lane);
#pragma unroll
        for (int j = 0; j < 2; ++j) bfrag[j] = ldfrag(Bb[cur], wn * 2 + j, lane);

#pragma unroll
        for (int i = 0; i < 4; ++i)
#pragma unroll
            for (int j = 0; j < 2; ++j)
                acc[i][j] = __builtin_amdgcn_wmma_f32_16x16x32_bf16(
                    false, af[i], false, bfrag[j], (short)0, acc[i][j],
                    false, false);
    }

    // Epilogue: add bias, write fp32 h. C layout: VGPR r -> row base+r (lanes
    // 0-15) / base+8+r (lanes 16-31); column = lane&15 within 16-wide tile.
#pragma unroll
    for (int j = 0; j < 2; ++j) {
        const int col = n0 + wn * 32 + j * 16 + (lane & 15);
        const float bv = bias[col];
#pragma unroll
        for (int i = 0; i < 4; ++i) {
            const int rb = m0 + wm * 64 + i * 16 + (lane >> 4) * 8;
#pragma unroll
            for (int r = 0; r < 8; ++r)
                h[(size_t)(rb + r) * D_HID + col] = acc[i][j][r] + bv;
        }
    }
}

// ------------------------- top-k mask + relu -------------------------------
// One 256-thread block per row. Exact k-th-largest via 4-pass radix select on
// order-preserving unsigned keys; 256-bin LDS histogram + parallel suffix scan.
__global__ __launch_bounds__(256)
void msae_topk_relu(float* __restrict__ out)
{
    const int row = blockIdx.x;
    const int t   = threadIdx.x;            // 0..255
    const int j0  = t * 16;
    const size_t SL = (size_t)N_ROWS * D_HID;

    union F16 { float4 q[4]; float f[16]; };

    // Load this thread's 16 columns of h (currently in slice 0).
    F16 u;
    const float4* src = (const float4*)(out + (size_t)row * D_HID + j0);
    u.q[0] = src[0]; u.q[1] = src[1]; u.q[2] = src[2]; u.q[3] = src[3];

    unsigned key[16];
#pragma unroll
    for (int i = 0; i < 16; ++i) {
        unsigned b = __float_as_uint(u.f[i]);
        key[i] = ((int)b < 0) ? ~b : (b ^ 0x80000000u);   // monotone desc map
    }

    __shared__ unsigned hist[256];
    __shared__ unsigned sc[256];
    __shared__ unsigned sPrefix, sRem;

    const unsigned ks[3] = {32u, 64u, 128u};
    unsigned T[3];

    for (int s = 0; s < 3; ++s) {
        unsigned prefix = 0, rem = ks[s];
#pragma unroll 1
        for (int shift = 24; shift >= 0; shift -= 8) {
            hist[t] = 0;
            __syncthreads();
            const unsigned hiMask =
                (shift == 24) ? 0u : (0xFFFFFFFFu << (shift + 8));
#pragma unroll
            for (int i = 0; i < 16; ++i)
                if ((key[i] & hiMask) == prefix)
                    atomicAdd(&hist[(key[i] >> shift) & 255u], 1u);
            __syncthreads();
            sc[t] = hist[t];
            __syncthreads();
            // suffix sum: sc[t] = sum(hist[t..255])
            for (int off = 1; off < 256; off <<= 1) {
                unsigned add = (t + off < 256) ? sc[t + off] : 0u;
                __syncthreads();
                sc[t] += add;
                __syncthreads();
            }
            const unsigned above = (t == 255) ? 0u : sc[t + 1];
            if (sc[t] >= rem && above < rem) {
                sPrefix = prefix | ((unsigned)t << shift);
                sRem    = rem - above;
            }
            __syncthreads();
            prefix = sPrefix;
            rem    = sRem;
            __syncthreads();
        }
        T[s] = prefix;   // k-th largest key; mask = (key >= T)
    }

    // Write 3 masked+relu slices (slice 0 overwritten in place).
#pragma unroll 1
    for (int s = 0; s < 3; ++s) {
        F16 o;
#pragma unroll
        for (int i = 0; i < 16; ++i) {
            const float v = u.f[i];
            o.f[i] = (key[i] >= T[s] && v > 0.0f) ? v : 0.0f;
        }
        float4* dst = (float4*)(out + (size_t)s * SL + (size_t)row * D_HID + j0);
        dst[0] = o.q[0]; dst[1] = o.q[1]; dst[2] = o.q[2]; dst[3] = o.q[3];
    }
}

// ------------------------------- launcher ----------------------------------
extern "C" void kernel_launch(void* const* d_in, const int* in_sizes, int n_in,
                              void* d_out, int out_size, void* d_ws,
                              size_t ws_size, hipStream_t stream)
{
    const float* x  = (const float*)d_in[0];   // [16384,1024]
    const float* W  = (const float*)d_in[1];   // [1024,4096]
    const float* bb = (const float*)d_in[2];   // [4096]
    float* out      = (float*)d_out;           // [3,16384,4096]

    dim3 g1(D_HID / BN, N_ROWS / BM);          // (32,128)
    msae_gemm_bf16<<<g1, 256, 0, stream>>>(x, W, bb, out);   // h -> slice 0
    msae_topk_relu<<<N_ROWS, 256, 0, stream>>>(out);
}
